// GraphSAGELayer_16518444220921
// MI455X (gfx1250) — matrix operand order repaired and checked by reading, not compiled, not run on hardware
//
#include <hip/hip_runtime.h>

typedef __bf16 bf16_t;
typedef __bf16 v8bf  __attribute__((ext_vector_type(8)));
typedef __bf16 v16bf __attribute__((ext_vector_type(16)));
typedef float  v8f   __attribute__((ext_vector_type(8)));
typedef float  v4f   __attribute__((ext_vector_type(4)));

#define N_NODES 500000
#define D 128
#define K2 256          // concat dim
#define HOPS 3
#define WPB 8           // waves per block
#define TILE_M 16       // nodes per wave-tile

// Low 32 bits of a generic shared-memory pointer == LDS byte address
// (aperture tag lives in addr[63:32]; FLAT/async LDS addressing truncates).
__device__ __forceinline__ unsigned lds_off(const void* p) {
    return (unsigned)(unsigned long long)p;
}

// One wave-wide async gather: 32 lanes x 16 B = one 512 B feature row -> LDS.
// Tracked by ASYNCcnt, zero VGPR staging.
__device__ __forceinline__ void async_row_load(unsigned ldsByte, const float* gsrc, int lane) {
    unsigned l = ldsByte + (unsigned)lane * 16u;
    unsigned long long g = (unsigned long long)(const void*)gsrc + (unsigned long long)lane * 16ull;
    asm volatile("global_load_async_to_lds_b128 %0, %1, off"
                 :: "v"(l), "v"(g) : "memory");
}

__global__ __launch_bounds__(256)
void sage_wmma_async(const float* __restrict__ feature,
                     const float* __restrict__ W,        // [256][128] fp32
                     const float* __restrict__ bias,     // [128]
                     const long long* __restrict__ nidx, // [HOPS][N] int64
                     float* __restrict__ out,
                     int nTiles)
{
    __shared__ bf16_t Wt[D][K2];                 // 64 KB  [n][k] transposed bf16 weights
    __shared__ float  Sst[WPB][TILE_M][D];       // 64 KB  running state, fp32
    __shared__ float  Nbf[WPB][2][TILE_M][D];    // 128 KB neighbor double buffer, fp32
                                                 // total 256 KB <= 320 KB/WGP

    const int tid  = threadIdx.x;
    const int wave = tid >> 5;
    const int lane = tid & 31;

    // cooperative: W fp32 [256][128] -> Wt bf16 [128][256] (transposed)
    for (int e = tid; e < K2 * D; e += 256) {
        const int k = e >> 7;
        const int n = e & 127;
        Wt[n][k] = (bf16_t)W[e];
    }
    __syncthreads();

    const int tile = blockIdx.x * WPB + wave;
    if (tile >= nTiles) return;                  // whole-wave guard: EXEC all-1s for WMMA
    const long long base = (long long)tile * TILE_M;

    float (*S)[D]  = Sst[wave];
    float (*N0)[D] = Nbf[wave][0];
    float (*N1)[D] = Nbf[wave][1];

    const int half = lane >> 4;                  // 0: lanes 0-15, 1: lanes 16-31
    const int l16  = lane & 15;

    float bfrag[8];
    #pragma unroll
    for (int nt = 0; nt < 8; ++nt) bfrag[nt] = bias[nt * 16 + l16];

    // ---- issue own, hop0 and hop1 gathers up front (in-order completion lets us pipeline) ----
    const unsigned sOff  = lds_off(&S[0][0]);
    const unsigned n0Off = lds_off(&N0[0][0]);
    const unsigned n1Off = lds_off(&N1[0][0]);

    #pragma unroll 4
    for (int r = 0; r < TILE_M; ++r)                     // own rows -> S      (16)
        async_row_load(sOff + (unsigned)r * 512u, feature + (base + r) * D, lane);
    {
        const long long* h = nidx + base;                // hop0 rows -> N0    (32)
        #pragma unroll 4
        for (int r = 0; r < TILE_M; ++r)
            async_row_load(n0Off + (unsigned)r * 512u, feature + h[r] * D, lane);
    }
    {
        const long long* h = nidx + (long long)N_NODES + base; // hop1 -> N1   (48)
        #pragma unroll 4
        for (int r = 0; r < TILE_M; ++r)
            async_row_load(n1Off + (unsigned)r * 512u, feature + h[r] * D, lane);
    }

    v8f acc[8];

    // [16 x 256] @ [256 x 128]: 8 K-steps x 8 N-tiles of v_wmma_f32_16x16x32_bf16.
    // A fragments: fp32 from LDS, converted to bf16 at fragment build.
    auto hop_compute = [&](const float (*NB)[D]) {
        #pragma unroll
        for (int nt = 0; nt < 8; ++nt) {
            v8f c;
            #pragma unroll
            for (int v = 0; v < 8; ++v) c[v] = bfrag[nt];
            acc[nt] = c;
        }
        #pragma unroll
        for (int kt = 0; kt < 8; ++kt) {
            const float* rowp = (kt < 4) ? &S[l16][0] : &NB[l16][0];
            const int kk = (kt & 3) * 32 + half * 8;     // half0: K{0..7,16..23}, half1: K{8..15,24..31}
            const v4f f0 = *(const v4f*)(rowp + kk);
            const v4f f1 = *(const v4f*)(rowp + kk + 4);
            const v4f f2 = *(const v4f*)(rowp + kk + 16);
            const v4f f3 = *(const v4f*)(rowp + kk + 20);
            v16bf av;
            #pragma unroll
            for (int i = 0; i < 4; ++i) {
                av[i]      = (bf16_t)f0[i];
                av[4 + i]  = (bf16_t)f1[i];
                av[8 + i]  = (bf16_t)f2[i];
                av[12 + i] = (bf16_t)f3[i];
            }
            const int kb = kt * 32 + half * 8;           // B: half0: K{0..7,16..23}, half1: K{8..15,24..31} at col N
            #pragma unroll
            for (int nt = 0; nt < 8; ++nt) {
                const bf16_t* wr = &Wt[nt * 16 + l16][kb];
                const v8bf b0 = *(const v8bf*)(wr);
                const v8bf b1 = *(const v8bf*)(wr + 16);
                v16bf bv;
                #pragma unroll
                for (int i = 0; i < 8; ++i) { bv[i] = b0[i]; bv[i + 8] = b1[i]; }
                acc[nt] = __builtin_amdgcn_wmma_f32_16x16x32_bf16(
                    false, av, false, bv, (short)0, acc[nt], false, false);
            }
        }
    };

    auto writeback_state = [&]() {                       // acc (f32) -> S, no conversion
        #pragma unroll
        for (int nt = 0; nt < 8; ++nt) {
            const int col = nt * 16 + l16;
            #pragma unroll
            for (int v = 0; v < 8; ++v) S[half * 8 + v][col] = acc[nt][v];
        }
    };

    // ---- hop 0: need S + N0. 48 issued, wait <=16 -> oldest 32 complete. ----
    asm volatile("s_wait_asynccnt 16" ::: "memory");
    hop_compute(N0);
    writeback_state();

    // N0 consumed -> refill with hop2 rows. Drain DS first (reads of N0 done).
    asm volatile("s_wait_dscnt 0" ::: "memory");
    {
        const long long* h = nidx + 2ll * N_NODES + base;
        #pragma unroll 4
        for (int r = 0; r < TILE_M; ++r)
            async_row_load(n0Off + (unsigned)r * 512u, feature + h[r] * D, lane);
    }

    // ---- hop 1: wait <=16 -> all of N1 (older) complete, hop2 fill may still fly ----
    asm volatile("s_wait_asynccnt 16" ::: "memory");
    hop_compute(N1);
    writeback_state();

    // ---- hop 2: wait for everything ----
    asm volatile("s_wait_asynccnt 0" ::: "memory");
    hop_compute(N0);

    // ---- epilogue: ReLU, stage f32 in S, coalesced 512 B row stores ----
    asm volatile("s_wait_dscnt 0" ::: "memory");
    #pragma unroll
    for (int nt = 0; nt < 8; ++nt) {
        const int col = nt * 16 + l16;
        #pragma unroll
        for (int v = 0; v < 8; ++v) {
            const float x = acc[nt][v];
            S[half * 8 + v][col] = x > 0.0f ? x : 0.0f;
        }
    }
    asm volatile("" ::: "memory");
    #pragma unroll 4
    for (int r = 0; r < TILE_M; ++r) {
        const v4f x = *(const v4f*)&S[r][lane * 4];
        *(v4f*)(out + (base + r) * D + lane * 4) = x;
    }
}

extern "C" void kernel_launch(void* const* d_in, const int* in_sizes, int n_in,
                              void* d_out, int out_size, void* d_ws, size_t ws_size,
                              hipStream_t stream) {
    (void)in_sizes; (void)n_in; (void)d_ws; (void)ws_size; (void)out_size;
    const float*     feature = (const float*)d_in[0];
    const float*     W       = (const float*)d_in[1];
    const float*     b       = (const float*)d_in[2];
    const long long* nidx    = (const long long*)d_in[3];
    float*           out     = (float*)d_out;

    const int nTiles = N_NODES / TILE_M;                 // 31250
    const int blocks = (nTiles + WPB - 1) / WPB;         // 3907
    sage_wmma_async<<<blocks, 256, 0, stream>>>(feature, W, b, nidx, out, nTiles);
}
